// MultiHeadRelationalModule_3616362463328
// MI455X (gfx1250) — matrix operand order repaired and checked by reading, not compiled, not run on hardware
//
#include <hip/hip_runtime.h>
#include <hip/hip_bf16.h>
#include <math.h>

// MI455X / gfx1250 fused MultiHeadRelationalModule.
// One workgroup (256 thr = 8 wave32) per batch sample; every GEMM on
// v_wmma_f32_16x16x32_f16 (f32 accum). Every stage computes X*W or its
// transpose W^T*X^T so the D tile stores as ONE contiguous 16B
// ds_store_b128 per lane. All guards are branchless selects; wave id is
// readfirstlane-uniform so tile loops are scalar.

typedef _Float16 h16v __attribute__((ext_vector_type(16)));
typedef _Float16 h8v  __attribute__((ext_vector_type(8)));
typedef _Float16 h4v  __attribute__((ext_vector_type(4)));
typedef float    f8v  __attribute__((ext_vector_type(8)));
typedef float    f4v  __attribute__((ext_vector_type(4)));

#define NT 256
#define EPSN 1e-5f

// ---------------- LDS layout (byte offsets; all 16B aligned) ----------------
#define OFF_PQ    0        // [64][200] f16  Q rows=node f, cols=192 (+pad)
#define OFF_PK    25600    // [64][200] f16  K
#define OFF_PVT   51200    // [192][72] f16  V^T: [n_global][node]
#define OFF_WQT   78848    // [64][72]  f16  q_lin_w^T [c][d]
#define OFF_WKT   88064    // [64][72]  f16  k_lin_w^T
#define OFF_WAT   97280    // [64][72]  f16  a_lin_w^T [n][c]
#define OFF_WL1T  106496   // [64][72]  f16  lin1 head slab^T [o][d]
#define OFF_S16   115712   // [64][72]  f16  S[f][c]
#define OFF_A16   124928   // [64][72]  f16  A[f][n]
#define OFF_E16   134144   // [64][72]  f16  E_h[f][d]
#define OFF_TOK   78848    // phase-A alias of WQT: tokens [64][32] f16
#define OFF_WPT   82944    // phase-A alias: proj weight^T [192][32] f16
#define OFF_H1    0        // phase-C alias of PQ: [64][72] f32
#define OFF_BIAS  143360   // f32[192] staged per-row bias table
#define OFF_STAT  144128   // f32: sum,sumsq,mean,rs,maxv[64]
#define SMEM_BYTES (144128 + 512)

// A fragment (16x32 f16) from row-major LDS [M][K] tile at (row0,col0).
__device__ __forceinline__ h16v lds_load_a(const _Float16* base, int ld,
                                           int row0, int col0, int lane) {
  const int hlf = lane >> 4;
  const _Float16* p = base + (row0 + (lane & 15)) * ld + col0 + 8 * hlf;
  h8v lo = *(const h8v*)(p);
  h8v hi = *(const h8v*)(p + 16);
  return __builtin_shufflevector(lo, hi, 0,1,2,3,4,5,6,7,8,9,10,11,12,13,14,15);
}

// B fragment (32x16 f16) from a row-major [N][K] buffer (i.e. B^T layout).
__device__ __forceinline__ h16v lds_load_bT(const _Float16* baseT, int ld,
                                            int col0, int krow0, int lane) {
  const int hlf = lane >> 4;
  const _Float16* p = baseT + (col0 + (lane & 15)) * ld + krow0 + 8 * hlf;
  h8v lo = *(const h8v*)(p);
  h8v hi = *(const h8v*)(p + 16);
  return __builtin_shufflevector(lo, hi, 0,1,2,3,4,5,6,7,8,9,10,11,12,13,14,15);
}

__device__ __forceinline__ f8v wmma16(h16v a, h16v b, f8v c) {
  return __builtin_amdgcn_wmma_f32_16x16x32_f16(false, a, false, b, (short)0, c,
                                                false, false);
}

__device__ __forceinline__ f8v fzero8() {
  f8v c;
#pragma unroll
  for (int e = 0; e < 8; ++e) c[e] = 0.f;
  return c;
}

// Store a D tile transposed: element e at (row=row0+e+8*hlf, col) goes to
// dstT[col*ld + row] -> contiguous in e -> ONE 16B ds_store_b128.
__device__ __forceinline__ void lds_store_dT(_Float16* dstT, int ld,
                                             int row0, int col0, int lane, h8v v) {
  const int hlf = lane >> 4;
  *(h8v*)(dstT + (col0 + (lane & 15)) * ld + row0 + 8 * hlf) = v;
}

__device__ __forceinline__ float elu1(float x) { return x > 0.f ? x : __expf(x) - 1.f; }

__global__ __launch_bounds__(NT, 1)
void mhrel_kernel(const float* __restrict__ x,
                  const float* __restrict__ c1w, const float* __restrict__ c1b,
                  const float* __restrict__ c2w, const float* __restrict__ c2b,
                  const float* __restrict__ kpw, const float* __restrict__ kpb,
                  const float* __restrict__ qpw, const float* __restrict__ qpb,
                  const float* __restrict__ vpw, const float* __restrict__ vpb,
                  const float* __restrict__ kng, const float* __restrict__ knb,
                  const float* __restrict__ qng, const float* __restrict__ qnb,
                  const float* __restrict__ vng, const float* __restrict__ vnb,
                  const float* __restrict__ klw, const float* __restrict__ klb,
                  const float* __restrict__ qlw, const float* __restrict__ qlb,
                  const float* __restrict__ alw, const float* __restrict__ alb,
                  const float* __restrict__ l1w, const float* __restrict__ l1b,
                  const float* __restrict__ l2w, const float* __restrict__ l2b,
                  float* __restrict__ out) {
  __shared__ __align__(16) unsigned char smem[SMEM_BYTES];
  const int tid  = threadIdx.x;
  const int lane = tid & 31;
  const int hlf  = lane >> 4;
  const int wvu  = __builtin_amdgcn_readfirstlane(tid >> 5);  // uniform wave id
  const int b    = blockIdx.x;

  _Float16* PQ   = (_Float16*)(smem + OFF_PQ);
  _Float16* PK   = (_Float16*)(smem + OFF_PK);
  _Float16* PVT  = (_Float16*)(smem + OFF_PVT);
  _Float16* WQT  = (_Float16*)(smem + OFF_WQT);
  _Float16* WKT  = (_Float16*)(smem + OFF_WKT);
  _Float16* WAT  = (_Float16*)(smem + OFF_WAT);
  _Float16* WL1T = (_Float16*)(smem + OFF_WL1T);
  _Float16* S16  = (_Float16*)(smem + OFF_S16);
  _Float16* A16  = (_Float16*)(smem + OFF_A16);
  _Float16* E16  = (_Float16*)(smem + OFF_E16);
  _Float16* TOK  = (_Float16*)(smem + OFF_TOK);
  _Float16* WPT  = (_Float16*)(smem + OFF_WPT);
  float*    BIAS = (float*)(smem + OFF_BIAS);
  float*    STAT = (float*)(smem + OFF_STAT);

  // ---------------- Phase A: tokens [49][22] (padded [64][32]) ----------------
  const float* xb = x + (size_t)b * 3 * 49;
  for (int i = tid; i < 64 * 32; i += NT) {
    int r = i >> 5, c = i & 31;
    if (r >= 49 || c >= 22) TOK[i] = (_Float16)0.f;
  }
  if (tid < 49) {
    const int p = tid;
    const float i0 = xb[p], i1 = xb[49 + p], i2 = xb[98 + p];
    float hc[16];
#pragma unroll
    for (int o = 0; o < 16; ++o)
      hc[o] = fmaxf(c1w[o*3+0]*i0 + c1w[o*3+1]*i1 + c1w[o*3+2]*i2 + c1b[o], 0.f);
    for (int o = 0; o < 20; ++o) {
      float acc = c2b[o];
#pragma unroll
      for (int i = 0; i < 16; ++i) acc += c2w[o*16+i]*hc[i];
      TOK[p*32 + o] = (_Float16)fmaxf(acc, 0.f);
    }
    TOK[p*32 + 20] = (_Float16)((float)(p % 7) / 7.0f);  // xc
    TOK[p*32 + 21] = (_Float16)((float)(p / 7) / 7.0f);  // yc
  }

  // ---------- Phase A2: K/Q/V projections + whole-tensor LN (t unrolled) ----------
  const float* PW[3] = {kpw, qpw, vpw};
  const float* PB[3] = {kpb, qpb, vpb};
  const float* NG[3] = {kng, qng, vng};
  const float* NB[3] = {knb, qnb, vnb};

#pragma unroll
  for (int t = 0; t < 3; ++t) {
    _Float16* Pt = (t == 1) ? PQ : PK;          // t==2 uses PVT
    for (int i = tid; i < 192 * 32; i += NT) {  // weight^T [n][k], k padded 22->32
      int n = i >> 5, k = i & 31;
      WPT[i] = (_Float16)((k < 22) ? PW[t][k*192 + n] : 0.f);
    }
    if (tid < 192) BIAS[tid] = PB[t][tid];
    if (tid < 2)   STAT[tid] = 0.f;
    __syncthreads();

    float ssum = 0.f, ssq = 0.f;
    if (t < 2) {
      // P^T = W^T(192x32) * tok^T(32x64): M=n (12 tiles), N=f (4 tiles)
#pragma unroll
      for (int jj = 0; jj < 6; ++jj) {
        const int j = wvu + 8*jj;
        const int mt = j >> 2, nt = j & 3;
        f8v c = wmma16(lds_load_a(WPT, 32, mt*16, 0, lane),
                       lds_load_bT(TOK, 32, nt*16, 0, lane), fzero8());
        const float sel = ((nt*16 + (lane & 15)) < 49) ? 1.f : 0.f;
        const f8v bias8 = *(const f8v*)(BIAS + mt*16 + 8*hlf);
        h8v pk;
#pragma unroll
        for (int e = 0; e < 8; ++e) {
          const float v = (c[e] + bias8[e]) * sel;
          ssum += v; ssq += v*v;
          pk[e] = (_Float16)v;
        }
        lds_store_dT(Pt, 200, mt*16, nt*16, lane, pk);   // Pt[f][n]
      }
    } else {
      // V = tok(64x32) * W(32x192): M=f (4 tiles), N=n (12 tiles); store V^T[n][f]
#pragma unroll
      for (int jj = 0; jj < 6; ++jj) {
        const int j = wvu + 8*jj;
        const int mt = j / 12, nt = j % 12;
        f8v c = wmma16(lds_load_a(TOK, 32, mt*16, 0, lane),
                       lds_load_bT(WPT, 32, nt*16, 0, lane), fzero8());
        const float bv = BIAS[nt*16 + (lane & 15)];
        h8v pk;
#pragma unroll
        for (int e = 0; e < 8; ++e) {
          const int f = mt*16 + e + 8*hlf;
          float v = c[e] + bv;
          v = (f < 49) ? v : 0.f;
          ssum += v; ssq += v*v;
          pk[e] = (_Float16)v;
        }
        lds_store_dT(PVT, 72, mt*16, nt*16, lane, pk);   // PVT[n][f]
      }
    }
    atomicAdd(&STAT[0], ssum);
    atomicAdd(&STAT[1], ssq);
    __syncthreads();
    if (tid == 0) {
      float mean = STAT[0] / 9408.f;
      float var  = STAT[1] / 9408.f - mean*mean;
      STAT[2] = mean; STAT[3] = rsqrtf(var + EPSN);
    }
    __syncthreads();
    const float mean = STAT[2], rs = STAT[3];
    if (t < 2) {
      // thread -> (row f, 48-col block); 12x h4v chunks, branchless clamp
      const int f  = tid >> 2;
      const int fc = (f < 49) ? f : 48;
      const float sel = (f < 49) ? 1.f : 0.f;
      const int n0 = (tid & 3) * 48;
#pragma unroll
      for (int nn = 0; nn < 48; nn += 4) {
        const int n  = n0 + nn;
        const int hh = n >> 6, d = n & 63;
        const int gi = (hh*49 + fc)*64 + d;
        h4v pv = *(const h4v*)(Pt + f*200 + n);
        f4v g4 = *(const f4v*)(NG[t] + gi);
        f4v b4 = *(const f4v*)(NB[t] + gi);
        h4v ov;
#pragma unroll
        for (int e = 0; e < 4; ++e)
          ov[e] = (_Float16)((((float)pv[e] - mean)*rs*g4[e] + b4[e]) * sel);
        *(h4v*)(Pt + f*200 + n) = ov;
      }
    } else {
      // thread -> column n of V^T; 8x h8v chunks along node f
      if (tid < 192) {
        const int n = tid, hh = n >> 6, d = n & 63;
        const float* gp = NG[2] + hh*49*64 + d;
        const float* bp = NB[2] + hh*49*64 + d;
#pragma unroll
        for (int f0 = 0; f0 < 64; f0 += 8) {
          h8v pv = *(const h8v*)(PVT + n*72 + f0);
          h8v ov;
#pragma unroll
          for (int e = 0; e < 8; ++e) {
            const int f  = f0 + e;
            const int fc = (f < 49) ? f : 48;
            float v = ((float)pv[e] - mean)*rs*gp[fc*64] + bp[fc*64];
            ov[e] = (_Float16)((f < 49) ? v : 0.f);
          }
          *(h8v*)(PVT + n*72 + f0) = ov;
        }
      }
    }
    __syncthreads();
  }

  // -------- Phase B: attention weights + bias tables (overwrite TOK/WPT) --------
  {
    const int r  = tid >> 2;          // 0..63
    const int k0 = (tid & 3) * 18;    // 72 = 4*18
    for (int kk = 0; kk < 18; ++kk) {
      const int k = k0 + kk;
      _Float16 vq = (_Float16)0.f, vk = (_Float16)0.f, va = (_Float16)0.f;
      if (k < 64 && r < 49) {
        vq = (_Float16)qlw[k*49 + r];
        vk = (_Float16)klw[k*49 + r];
        if (k < 49) va = (_Float16)alw[k*49 + r];
      }
      WQT[r*72 + k] = vq; WKT[r*72 + k] = vk; WAT[r*72 + k] = va;
    }
  }
  if (tid < 64) {
    BIAS[tid]      = (tid < 49) ? (qlb[tid] + klb[tid]) : 0.f;  // S bias per c
    BIAS[64 + tid] = (tid < 49) ? alb[tid] : 0.f;               // A bias per n
  }

  f8v h1acc[2];
  h1acc[0] = fzero8(); h1acc[1] = fzero8();

  for (int h = 0; h < 3; ++h) {
    {
      const int r  = tid >> 2;
      const int k0 = (tid & 3) * 18;
      for (int kk = 0; kk < 18; ++kk) {
        const int k = k0 + kk;
        WL1T[r*72 + k] = (_Float16)((k < 64) ? l1w[(h*64 + k)*64 + r] : 0.f);
      }
    }
    __syncthreads();

    // step 1: S^T = WQT(c,d)*Q^T + WKT*K^T -> elu -> S16[f][c]
#pragma unroll
    for (int u = 0; u < 2; ++u) {
      const int j = wvu + 8*u, mt = j >> 2, nt = j & 3;  // mt: c tile, nt: f tile
      f8v c = fzero8();
#pragma unroll
      for (int ks = 0; ks < 2; ++ks) {
        c = wmma16(lds_load_a(WQT, 72, mt*16, ks*32, lane),
                   lds_load_bT(PQ, 200, nt*16, h*64 + ks*32, lane), c);
        c = wmma16(lds_load_a(WKT, 72, mt*16, ks*32, lane),
                   lds_load_bT(PK, 200, nt*16, h*64 + ks*32, lane), c);
      }
      const f8v bias8 = *(const f8v*)(BIAS + mt*16 + 8*hlf);
      h8v pk;
#pragma unroll
      for (int e = 0; e < 8; ++e) pk[e] = (_Float16)elu1(c[e] + bias8[e]);
      lds_store_dT(S16, 72, mt*16, nt*16, lane, pk);
    }
    __syncthreads();

    // step 2: A^T = WAT(n,c)*S^T -> A16[f][n]
#pragma unroll
    for (int u = 0; u < 2; ++u) {
      const int j = wvu + 8*u, mt = j >> 2, nt = j & 3;  // mt: n tile, nt: f tile
      f8v c = fzero8();
#pragma unroll
      for (int ks = 0; ks < 2; ++ks)
        c = wmma16(lds_load_a(WAT, 72, mt*16, ks*32, lane),
                   lds_load_bT(S16, 72, nt*16, ks*32, lane), c);
      const f8v bias8 = *(const f8v*)(BIAS + 64 + mt*16 + 8*hlf);
      h8v pk;
#pragma unroll
      for (int e = 0; e < 8; ++e) pk[e] = (_Float16)(c[e] + bias8[e]);
      lds_store_dT(A16, 72, mt*16, nt*16, lane, pk);
    }
    __syncthreads();

    // step 3: row softmax over n<49 (vectorized row in registers)
    if (tid < 49) {
      _Float16* rowp = A16 + tid*72;
      float vb[56];
#pragma unroll
      for (int ch = 0; ch < 7; ++ch) {
        h8v hv = *(const h8v*)(rowp + ch*8);
#pragma unroll
        for (int e = 0; e < 8; ++e) vb[ch*8 + e] = (float)hv[e];
      }
      float m = -1e30f;
#pragma unroll
      for (int i = 0; i < 49; ++i) m = fmaxf(m, vb[i]);
      float s = 0.f;
#pragma unroll
      for (int i = 0; i < 49; ++i) { vb[i] = __expf(vb[i] - m); s += vb[i]; }
      const float inv = 1.f / s;
#pragma unroll
      for (int ch = 0; ch < 7; ++ch) {
        h8v hv;
#pragma unroll
        for (int e = 0; e < 8; ++e) {
          const int i = ch*8 + e;
          hv[e] = (_Float16)((i < 49) ? vb[i]*inv : 0.f);
        }
        *(h8v*)(rowp + ch*8) = hv;
      }
    }
    __syncthreads();

    // step 4: E^T = V^T(d,n)*A^T -> E16[f][d]
#pragma unroll
    for (int u = 0; u < 2; ++u) {
      const int j = wvu + 8*u, mt = j >> 2, nt = j & 3;  // mt: d tile, nt: f tile
      f8v c = fzero8();
#pragma unroll
      for (int ks = 0; ks < 2; ++ks)
        c = wmma16(lds_load_a(PVT, 72, h*64 + mt*16, ks*32, lane),
                   lds_load_bT(A16, 72, nt*16, ks*32, lane), c);
      h8v pk;
#pragma unroll
      for (int e = 0; e < 8; ++e) pk[e] = (_Float16)c[e];
      lds_store_dT(E16, 72, mt*16, nt*16, lane, pk);
    }
    __syncthreads();

    // fused lin1: H1 += E_h(f,d) @ W1slab(d,o)  (persistent accumulators)
#pragma unroll
    for (int u = 0; u < 2; ++u) {
      const int j = wvu + 8*u, mt = j >> 2, nt = j & 3;  // mt: f tile, nt: o tile
#pragma unroll
      for (int ks = 0; ks < 2; ++ks)
        h1acc[u] = wmma16(lds_load_a(E16, 72, mt*16, ks*32, lane),
                          lds_load_bT(WL1T, 72, nt*16, ks*32, lane), h1acc[u]);
    }
    __syncthreads();
  }

  // ---------------- Phase C: bias+relu, LN(49x64), col-max, lin2+elu -------------
  if (tid < 2) STAT[tid] = 0.f;
  __syncthreads();
  float* H1 = (float*)(smem + OFF_H1);   // aliases PQ (dead)
  float ssum = 0.f, ssq = 0.f;
#pragma unroll
  for (int u = 0; u < 2; ++u) {
    const int j = wvu + 8*u, mt = j >> 2, nt = j & 3;
    const int col  = nt*16 + (lane & 15);
    const float bv = l1b[col];
#pragma unroll
    for (int e = 0; e < 8; ++e) {
      const int row = mt*16 + e + 8*hlf;
      const float v  = fmaxf(h1acc[u][e] + bv, 0.f);
      const float vm = (row < 49) ? v : 0.f;
      ssum += vm; ssq += vm*vm;
      H1[row*72 + col] = v;
    }
  }
  atomicAdd(&STAT[0], ssum);
  atomicAdd(&STAT[1], ssq);
  __syncthreads();
  if (tid == 0) {
    float mean = STAT[0] / 3136.f;
    float var  = STAT[1] / 3136.f - mean*mean;
    STAT[2] = mean; STAT[3] = rsqrtf(var + EPSN);
  }
  __syncthreads();
  if (tid < 64) {
    const float mean = STAT[2], rs = STAT[3];
    float m = -1e30f;
    for (int r = 0; r < 49; ++r) m = fmaxf(m, (H1[r*72 + tid] - mean)*rs);
    STAT[4 + tid] = m;
  }
  __syncthreads();
  if (tid < 5) {
    float acc = l2b[tid];
#pragma unroll
    for (int d = 0; d < 64; ++d) acc += STAT[4 + d] * l2w[d*5 + tid];
    out[(size_t)b*5 + tid] = elu1(acc);
  }
}

extern "C" void kernel_launch(void* const* d_in, const int* in_sizes, int n_in,
                              void* d_out, int out_size, void* d_ws, size_t ws_size,
                              hipStream_t stream) {
  (void)n_in; (void)out_size; (void)d_ws; (void)ws_size;
  const float* p[27];
  for (int i = 0; i < 27; ++i) p[i] = (const float*)d_in[i];
  const int Bn = in_sizes[0] / (3 * 7 * 7);
  hipLaunchKernelGGL(mhrel_kernel, dim3(Bn), dim3(NT), 0, stream,
      p[0],  p[1],  p[2],  p[3],  p[4],
      p[5],  p[6],  p[7],  p[8],  p[9],  p[10],
      p[11], p[12], p[13], p[14], p[15], p[16],
      p[17], p[18], p[19], p[20], p[21], p[22],
      p[23], p[24], p[25], p[26],
      (float*)d_out);
}
